// MinConv2dGRU_33526514713038
// MI455X (gfx1250) — matrix operand order repaired and checked by reading, not compiled, not run on hardware
//
#include <hip/hip_runtime.h>
#include <hip/hip_bf16.h>

// ---------------------------------------------------------------------------
// MinConv2dGRU on MI455X (gfx1250, wave32):
//   conv(3x3, pad1) as implicit GEMM on v_wmma_f32_16x16x32_f16 (f32 accum),
//   im2col staged once per block into LDS in WMMA B-fragment order,
//   halo slab filled via GLOBAL_LOAD_ASYNC_TO_LDS_B128 (ASYNCcnt-tracked),
//   minGRU scan as a bandwidth-bound sequential pass over S.
// Shapes: B=4 S=64 Cin=32 H=W=32; per layer Cout=64 (conv emits 2*64 = 128 ch).
// ---------------------------------------------------------------------------

typedef __attribute__((ext_vector_type(16))) _Float16 v16h;
typedef __attribute__((ext_vector_type(8)))  float    v8f;
typedef int v4i __attribute__((vector_size(16)));

#define NIMG   256          // B*S images per layer
#define HW     1024         // 32*32 pixels
#define GHCH   128          // conv output channels (gate+hidden)
#define OUTCH  64           // hidden channels per layer

#if defined(__HIP_DEVICE_COMPILE__) && \
    __has_builtin(__builtin_amdgcn_global_load_async_to_lds_b128) && \
    __has_builtin(__builtin_amdgcn_s_wait_asynccnt)
#define USE_ASYNC_LDS 1
#else
#define USE_ASYNC_LDS 0
#endif

#if USE_ASYNC_LDS
// One per-lane 16B async copy, global -> LDS (tracked by ASYNCcnt).
__device__ __forceinline__ void async_copy16(const _Float16* gsrc, _Float16* ldst) {
    __builtin_amdgcn_global_load_async_to_lds_b128(
        (__attribute__((address_space(1))) v4i*)const_cast<_Float16*>(gsrc),
        (__attribute__((address_space(3))) v4i*)ldst,
        0, 0);
}
#endif

// ---------------------------------------------------------------------------
// Pack conv weights (O=128, I=CIN, 3, 3) f32 -> f16 in the exact per-lane
// A-fragment layout of V_WMMA_F32_16X16X32_F16:
//   A 16x32 (MxK): lanes 0-15: K = {0..7, 16..23}; lanes 16-31: K = {8..15, 24..31}
// dst[((mtile*NK + kchunk)*32 + lane)*16 + e]
// ---------------------------------------------------------------------------
template <int CIN>
__global__ void pack_weights(const float* __restrict__ w, _Float16* __restrict__ dst) {
    constexpr int NK = (CIN * 9) / 32;     // K = CIN*9 is a multiple of 32 here
    const int total = 8 * NK * 32 * 16;    // 8 M-tiles of 16 rows
    int id = blockIdx.x * blockDim.x + threadIdx.x;
    if (id >= total) return;
    int e    = id & 15;
    int lane = (id >> 4) & 31;
    int mtkc = id >> 9;
    int kc   = mtkc % NK;
    int mt   = mtkc / NK;
    int m    = mt * 16 + (lane & 15);
    int klocal = ((lane & 16) ? 8 : 0) + (e < 8 ? e : e + 8);
    int k  = kc * 32 + klocal;
    int ci = k / 9;
    int t  = k - ci * 9;
    int dy = t / 3;
    int dx = t - dy * 3;
    dst[id] = (_Float16)w[((m * CIN + ci) * 3 + dy) * 3 + dx];
}

// ---------------------------------------------------------------------------
// f32 -> f16 elementwise (input image conversion)
// ---------------------------------------------------------------------------
__global__ void f32_to_f16(const float* __restrict__ src, _Float16* __restrict__ dst, int n) {
    int i = blockIdx.x * blockDim.x + threadIdx.x;
    if (i < n) dst[i] = (_Float16)src[i];
}

// ---------------------------------------------------------------------------
// Implicit-GEMM conv row kernel.
// Grid: NIMG*32 workgroups of 128 threads (4 waves). Block = (image n, row h).
//
// Stage 1: halo slab in LDS, rows h-1..h+1 of all CIN channels, padded row
//          stride 48 f16 (96 B): col 7 = left halo, 8..39 = data (16B-aligned
//          async b128 destinations), 40 = right halo.
// Stage 2: im2col ONCE per block into bpack[], stored in the exact per-lane
//          WMMA B-fragment order: bpack[((kc*2+nt)*32+lane)*16+e] where
//          k = kc*32 + (lane>=16 ? 8:0) + (e<8 ? e : e+8), N = nt*16 + (lane&15).
// Stage 3: GEMM M=128, N=32, K=CIN*9. Wave wv owns M-tiles {2wv,2wv+1} x
//          N-tiles {0,1}: per K-step 2 A frags (global b128 x2 each),
//          2 B frags (ds_load_b128 x2 each), 4 WMMAs.
// ---------------------------------------------------------------------------
template <int CIN>
__global__ __launch_bounds__(128)
void conv_gru_wmma(const _Float16* __restrict__ in,
                   const _Float16* __restrict__ wpack,
                   const float*    __restrict__ bias,
                   float*          __restrict__ gh) {
    constexpr int NK      = (CIN * 9) / 32;
    constexpr int ROWSETS = CIN * 3;                 // (channel, halo-row) pairs
    __shared__ __attribute__((aligned(16))) _Float16 slab[ROWSETS * 48];
    __shared__ __attribute__((aligned(16))) _Float16 bpack[NK * 1024];

    const int n   = blockIdx.x >> 5;
    const int h   = blockIdx.x & 31;
    const int tid = threadIdx.x;

    // ---- Stage 1: halo slab fill --------------------------------------------
#if USE_ASYNC_LDS
    // Zero-pad halo columns / out-of-image rows (addresses disjoint from the
    // async destinations, so no ds/async ordering hazard).
    for (int i = tid; i < ROWSETS; i += 128) {
        int r    = i % 3;
        int srow = h + r - 1;
        _Float16* row = &slab[i * 48];
        if (srow >= 0 && srow < 32) {
            row[7]  = (_Float16)0.0f;
            row[40] = (_Float16)0.0f;
        } else {
            for (int sc = 7; sc <= 40; ++sc) row[sc] = (_Float16)0.0f;
        }
    }
    // Interior: one async b128 per 8 f16; per-lane src/dst addresses.
    for (int i = tid; i < ROWSETS * 4; i += 128) {
        int rs = i >> 2, q = i & 3;
        int ci = rs / 3, r = rs % 3;
        int srow = h + r - 1;
        if (srow >= 0 && srow < 32) {
            const _Float16* src = in + ((size_t)(n * CIN + ci) * 32 + srow) * 32 + q * 8;
            async_copy16(src, &slab[rs * 48 + 8 + q * 8]);
        }
    }
    __builtin_amdgcn_s_wait_asynccnt(0);
#else
    for (int i = tid; i < ROWSETS * 34; i += 128) {
        int rs = i / 34;
        int sc = i - rs * 34 + 7;        // slab col 7..40
        int r  = rs % 3;
        int srow = h + r - 1;
        int scol = sc - 8;               // input col (-1..32)
        _Float16 v = (_Float16)0.0f;
        if (srow >= 0 && srow < 32 && scol >= 0 && scol < 32) {
            int ci = rs / 3;
            v = in[((size_t)(n * CIN + ci) * 32 + srow) * 32 + scol];
        }
        slab[rs * 48 + sc] = v;
    }
#endif
    __syncthreads();

    // ---- Stage 2: im2col into B-fragment order (done once, shared by waves) --
    for (int i = tid; i < NK * 1024; i += 128) {
        int e    = i & 15;
        int lane = (i >> 4) & 31;
        int nt   = (i >> 9) & 1;
        int kc   = i >> 10;
        int k    = kc * 32 + ((lane & 16) ? 8 : 0) + (e < 8 ? e : e + 8);
        int ci   = k / 9;
        int t    = k - ci * 9;
        int dy   = t / 3;
        int dx   = t - dy * 3;
        int npx  = nt * 16 + (lane & 15);
        bpack[i] = slab[(ci * 3 + dy) * 48 + 7 + npx + dx];
    }
    __syncthreads();

    // ---- Stage 3: WMMA GEMM --------------------------------------------------
    const int lane = tid & 31;
    const int wv   = tid >> 5;
    const int mt0  = wv * 2;
    const int mt1  = wv * 2 + 1;
    const int npix = lane & 15;

    v8f acc00 = {}, acc01 = {}, acc10 = {}, acc11 = {};

    for (int kc = 0; kc < NK; ++kc) {
        v16h a0 = *(const v16h*)(wpack + ((size_t)(mt0 * NK + kc) * 32 + lane) * 16);
        v16h a1 = *(const v16h*)(wpack + ((size_t)(mt1 * NK + kc) * 32 + lane) * 16);
        v16h b0 = *(const v16h*)(bpack + (size_t)kc * 1024 + lane * 16);
        v16h b1 = *(const v16h*)(bpack + (size_t)kc * 1024 + 512 + lane * 16);

        acc00 = __builtin_amdgcn_wmma_f32_16x16x32_f16(false, a0, false, b0, (short)0, acc00, false, false);
        acc01 = __builtin_amdgcn_wmma_f32_16x16x32_f16(false, a0, false, b1, (short)0, acc01, false, false);
        acc10 = __builtin_amdgcn_wmma_f32_16x16x32_f16(false, a1, false, b0, (short)0, acc10, false, false);
        acc11 = __builtin_amdgcn_wmma_f32_16x16x32_f16(false, a1, false, b1, (short)0, acc11, false, false);
    }

    // C/D layout: VGPR r -> M = r (lanes 0-15) / r+8 (lanes 16-31), N = lane&15.
    const int mrow = (lane & 16) ? 8 : 0;
    float* ghn = gh + (size_t)n * (GHCH * HW) + h * 32;
#pragma unroll
    for (int r = 0; r < 8; ++r) {
        int c0 = mt0 * 16 + r + mrow;
        int c1 = mt1 * 16 + r + mrow;
        ghn[(size_t)c0 * HW + npix]      = acc00[r] + bias[c0];
        ghn[(size_t)c0 * HW + 16 + npix] = acc01[r] + bias[c0];
        ghn[(size_t)c1 * HW + npix]      = acc10[r] + bias[c1];
        ghn[(size_t)c1 * HW + 16 + npix] = acc11[r] + bias[c1];
    }
}

// ---------------------------------------------------------------------------
// minGRU scan along S. One thread per (b, c, h, w); gh holds 128 channels:
// gate = ch[0..63], hidden = ch[64..127]. h_t = (1-z)h_{t-1} + z*g(hidden).
// ---------------------------------------------------------------------------
__global__ void mingru_scan(const float* __restrict__ gh,
                            float*    __restrict__ out_f32,   // may be null
                            _Float16* __restrict__ out_f16,   // may be null
                            float*    __restrict__ hlast) {
    int id = blockIdx.x * blockDim.x + threadIdx.x;   // 4*64*1024 = 262144 threads
    int hw = id & (HW - 1);
    int c  = (id >> 10) & (OUTCH - 1);
    int b  = id >> 16;

    const float* gp = gh + ((size_t)(b * 64) * GHCH + c) * HW + hw;
    const float* hp = gp + (size_t)OUTCH * HW;
    const size_t sstride = (size_t)GHCH * HW;

    float hstate = 0.5f;   // g(0) = 0.5
    for (int s = 0; s < 64; ++s) {
        float gate = gp[s * sstride];
        float hid  = hp[s * sstride];
        if (s + 1 < 64) {  // -> global_prefetch_b8 for the next timestep
            __builtin_prefetch(&gp[(s + 1) * sstride], 0, 0);
            __builtin_prefetch(&hp[(s + 1) * sstride], 0, 0);
        }
        float z  = 1.0f / (1.0f + __expf(-gate));
        float gv = (hid >= 0.0f) ? (hid + 0.5f) : (1.0f / (1.0f + __expf(-hid)));
        hstate = (1.0f - z) * hstate + z * gv;

        size_t oidx = ((size_t)(b * 64 + s) * OUTCH + c) * HW + hw;
        if (out_f32) out_f32[oidx] = hstate;
        if (out_f16) out_f16[oidx] = (_Float16)hstate;
    }
    hlast[((size_t)b * OUTCH + c) * HW + hw] = hstate;
}

// ---------------------------------------------------------------------------
// Host-side launch. Inputs: x, w1, b1, w2, b2 (all f32).
// d_out = [ out2 (16,777,216 f32) | h1 (262,144) | h2 (262,144) ].
// ---------------------------------------------------------------------------
extern "C" void kernel_launch(void* const* d_in, const int* in_sizes, int n_in,
                              void* d_out, int out_size, void* d_ws, size_t ws_size,
                              hipStream_t stream) {
    (void)n_in; (void)out_size; (void)ws_size;

    const float* x  = (const float*)d_in[0];
    const float* w1 = (const float*)d_in[1];
    const float* b1 = (const float*)d_in[2];
    const float* w2 = (const float*)d_in[3];
    const float* b2 = (const float*)d_in[4];
    float* out = (float*)d_out;

    const int xN = in_sizes[0];                   // 4*64*32*32*32 = 8,388,608

    // Workspace layout (bytes, aligned regions).
    char* ws = (char*)d_ws;
    _Float16* wp1  = (_Float16*)(ws);                               // 73,728 B
    _Float16* wp2  = (_Float16*)(ws + (256u << 10));                // 147,456 B
    _Float16* xh   = (_Float16*)(ws + (1u   << 20));                // 16.78 MB
    _Float16* o1h  = (_Float16*)(ws + (1u   << 20) + (34u << 20));  // 33.55 MB
    float*    gh   = (float*)   (ws + (1u   << 20) + (68u << 20));  // 134.2 MB

    float* h1 = out + (size_t)NIMG * OUTCH * HW;          // 16,777,216
    float* h2 = h1  + (size_t)4 * OUTCH * HW;             // +262,144

    // Weight packing + input conversion.
    pack_weights<32><<<(8 * 9  * 512 + 255) / 256, 256, 0, stream>>>(w1, wp1);
    pack_weights<64><<<(8 * 18 * 512 + 255) / 256, 256, 0, stream>>>(w2, wp2);
    f32_to_f16<<<(xN + 255) / 256, 256, 0, stream>>>(x, xh, xN);

    // Layer 1: conv (WMMA) -> scan (writes f16 sequence for layer 2 + h1).
    conv_gru_wmma<32><<<NIMG * 32, 128, 0, stream>>>(xh, wp1, b1, gh);
    mingru_scan<<<(4 * OUTCH * HW) / 256, 256, 0, stream>>>(gh, nullptr, o1h, h1);

    // Layer 2: conv (WMMA) -> scan (writes out2 f32 + h2).
    conv_gru_wmma<64><<<NIMG * 32, 128, 0, stream>>>(o1h, wp2, b2, gh);
    mingru_scan<<<(4 * OUTCH * HW) / 256, 256, 0, stream>>>(gh, out, nullptr, h2);
}
